// Qwen3VLVisionAttention_33809982554752
// MI455X (gfx1250) — matrix operand order repaired and checked by reading, not compiled, not run on hardware
//
#include <hip/hip_runtime.h>

// ---------------------------------------------------------------------------
// Qwen3-VL vision attention block for MI455X (gfx1250), wave32 + WMMA.
// B=4, S=1024, DIM=1280, H=16, HD=80.  All matmuls via v_wmma_f32_16x16x32_f16.
// GEMMs: 128x128 block, 8 waves, 32x64 tile per wave, double-buffered LDS.
// ---------------------------------------------------------------------------

typedef __attribute__((ext_vector_type(16))) _Float16 v16h;
typedef __attribute__((ext_vector_type(8)))  _Float16 v8h;
typedef __attribute__((ext_vector_type(8)))  float    v8f;

#define DEV_INLINE __device__ __forceinline__

static constexpr int Bc   = 4;
static constexpr int Sc   = 1024;
static constexpr int DIMc = 1280;
static constexpr int Hc   = 16;
static constexpr int HDc  = 80;
static constexpr int HDP  = 96;          // HD padded to 3*32 for WMMA K-steps
static constexpr float SCALE_C = 0.11180339887498948f;   // 80^-0.5

// ---- WMMA helpers ---------------------------------------------------------

DEV_INLINE v8f wmma_f16(v16h a, v16h b, v8f c) {
  // v_wmma_f32_16x16x32_f16: D = A(16x32 f16) * B(32x16 f16) + C(16x16 f32)
  return __builtin_amdgcn_wmma_f32_16x16x32_f16(
      /*neg_a=*/false, a, /*neg_b=*/false, b,
      /*c_mod=*/(short)0, c, /*reuse_a=*/false, /*reuse_b=*/false);
}

// A fragment (16xK tile, row-major, f16):
// lane = m + 16*hi, element i holds k = 16*hi + i (matches load_b_frag).
DEV_INLINE v16h load_a_frag(const _Float16* base, int row_stride, int lane) {
  int m = lane & 15, hi = lane >> 4;
  return *(const v16h*)(base + (size_t)m * row_stride + 16 * hi);
}

// B fragment from a B^T layout (rows = n, cols = k, f16):
// lane holds column n = lane&15, k = 16*(lane>>4) + i  -> one contiguous 32B load.
DEV_INLINE v16h load_b_frag(const _Float16* base, int row_stride, int lane) {
  int n = lane & 15, hi = lane >> 4;
  return *(const v16h*)(base + (size_t)n * row_stride + 16 * hi);
}

// ---- prep kernels ---------------------------------------------------------

__global__ void cvt_f32_to_f16_kernel(const float* __restrict__ src,
                                      _Float16* __restrict__ dst, size_t n) {
  size_t i = (size_t)blockIdx.x * blockDim.x + threadIdx.x;
  if (i < n) dst[i] = (_Float16)src[i];
}

// Wt[n*K + k] = W[k*N + n]  (f32 -> f16), writes coalesced along k.
__global__ void transpose_cvt_kernel(const float* __restrict__ W,
                                     _Float16* __restrict__ Wt, int K, int N) {
  size_t i = (size_t)blockIdx.x * blockDim.x + threadIdx.x;
  if (i >= (size_t)K * N) return;
  int k = (int)(i % K);
  int n = (int)(i / K);
  Wt[i] = (_Float16)W[(size_t)k * N + n];
}

// ---- tiled WMMA GEMM:  C[M][N] = A[M][K](f16) * Bt[N][K]^T(f16) + bias ----
// 128x128 block, BK=64, 8 waves (4 M x 2 N), 32x64 tile per wave,
// double-buffered LDS staging (global_load_b128 -> ds_store_b128).

__global__ void __launch_bounds__(256)
gemm_f16_bias_kernel(const _Float16* __restrict__ A,   // [M][K] f16 row-major
                     const _Float16* __restrict__ Bt,  // [N][K] f16 (B transposed)
                     const float*    __restrict__ bias,// [N] (may be null)
                     float*          __restrict__ C,   // [M][N] f32
                     int M, int N, int K) {
  constexpr int BM = 128, BN = 128, BK = 64;
  __shared__ _Float16 As[2][BM * BK];   // 2 x 16 KB
  __shared__ _Float16 Bs[2][BN * BK];   // 2 x 16 KB   (stored [n][k])

  const int tid  = threadIdx.x;
  const int lane = tid & 31;
  const int wave = tid >> 5;           // 8 waves
  const int wm   = wave & 3;           // 4 M-tiles of 32
  const int wn   = wave >> 2;          // 2 N-tiles of 64
  const int m0   = blockIdx.x * BM;
  const int n0   = blockIdx.y * BN;

  const int row = tid >> 1, cc = tid & 1;   // staging role: 32 f16 per thread/tile

  v8f acc[2][4] = {};                  // [a-16-tile][b-16-tile]

  auto stage = [&](int buf, int k0) {
    { // A tile: 128 rows x 64 k, 32 f16 (two v16h) per thread
      const v16h* g = (const v16h*)(A + (size_t)(m0 + row) * K + k0 + 32 * cc);
      v16h* l = (v16h*)(&As[buf][row * BK + 32 * cc]);
      l[0] = g[0];
      l[1] = g[1];
    }
    { // B^T tile: 128 rows x 64 k
      const v16h* g = (const v16h*)(Bt + (size_t)(n0 + row) * K + k0 + 32 * cc);
      v16h* l = (v16h*)(&Bs[buf][row * BK + 32 * cc]);
      l[0] = g[0];
      l[1] = g[1];
    }
  };

  const int nsteps = K / BK;
  stage(0, 0);
  __syncthreads();

  for (int step = 0; step < nsteps; ++step) {
    const int cur = step & 1;
    if (step + 1 < nsteps) stage(cur ^ 1, (step + 1) * BK);

    const _Float16* sA = As[cur];
    const _Float16* sB = Bs[cur];
#pragma unroll
    for (int kk = 0; kk < BK; kk += 32) {
      v16h a0 = load_a_frag(sA + (wm * 32 + 0)  * BK + kk, BK, lane);
      v16h a1 = load_a_frag(sA + (wm * 32 + 16) * BK + kk, BK, lane);
#pragma unroll
      for (int bn = 0; bn < 4; ++bn) {
        v16h bf = load_b_frag(sB + (wn * 64 + bn * 16) * BK + kk, BK, lane);
        acc[0][bn] = wmma_f16(a0, bf, acc[0][bn]);
        acc[1][bn] = wmma_f16(a1, bf, acc[1][bn]);
      }
    }
    __syncthreads();   // next buffer fully in LDS, current fully consumed
  }

  // Epilogue: C/D layout -> lane n = lane&15, row m = r + 8*(lane>>4)
  const int hi = lane >> 4, nn = lane & 15;
#pragma unroll
  for (int a = 0; a < 2; ++a)
#pragma unroll
    for (int bn = 0; bn < 4; ++bn) {
      int n = n0 + wn * 64 + bn * 16 + nn;
      float bv = bias ? bias[n] : 0.0f;
      v8f v = acc[a][bn];
#pragma unroll
      for (int r = 0; r < 8; ++r) {
        int m = m0 + wm * 32 + a * 16 + r + 8 * hi;
        C[(size_t)m * N + n] = v[r] + bv;
      }
    }
}

// ---- RoPE + repack kernel -------------------------------------------------
// qkv f32 [B,S,3,H,80] -> Qp,Kp f16 [B,H,S,96] (rope applied, zero-padded),
//                          Vt f16 [B,H,80,S] (transposed for PV B-fragments).

__global__ void rope_pack_kernel(const float* __restrict__ qkv,
                                 const float* __restrict__ cosb,  // [S][80]
                                 const float* __restrict__ sinb,  // [S][80]
                                 _Float16* __restrict__ Qp,
                                 _Float16* __restrict__ Kp,
                                 _Float16* __restrict__ Vt) {
  size_t idx = (size_t)blockIdx.x * blockDim.x + threadIdx.x;
  if (idx >= (size_t)Bc * Hc * Sc * HDP) return;
  int d = (int)(idx % HDP);
  size_t t = idx / HDP;
  int s = (int)(t % Sc); t /= Sc;
  int h = (int)(t % Hc);
  int b = (int)(t / Hc);

  size_t qo = ((size_t)(b * Hc + h) * Sc + s) * HDP + d;
  if (d >= HDc) {  // zero pad so the 96-wide dot product equals the 80-wide one
    Qp[qo] = (_Float16)0.f;
    Kp[qo] = (_Float16)0.f;
    return;
  }
  size_t base = ((size_t)b * Sc + s) * (3 * DIMc) + (size_t)h * HDc + d;
  float q = qkv[base];
  float k = qkv[base + DIMc];
  float v = qkv[base + 2 * DIMc];
  int doff = (d < 40) ? 40 : -40;
  float sgn = (d < 40) ? -1.0f : 1.0f;
  float qr = sgn * qkv[base + doff];
  float kr = sgn * qkv[base + DIMc + doff];
  float c  = cosb[s * HDc + d];
  float sn = sinb[s * HDc + d];
  Qp[qo] = (_Float16)(q * c + qr * sn);
  Kp[qo] = (_Float16)(k * c + kr * sn);
  Vt[((size_t)(b * Hc + h) * HDc + d) * Sc + s] = (_Float16)v;
}

// ---- flash attention kernel ----------------------------------------------
// grid (S/64, H, B), 128 threads = 4 waves, 16 query rows per wave.

__global__ void __launch_bounds__(128)
attn_kernel(const _Float16* __restrict__ Qp,  // [B,H,S,96]
            const _Float16* __restrict__ Kp,  // [B,H,S,96]
            const _Float16* __restrict__ Vt,  // [B,H,80,S]
            _Float16*       __restrict__ Oh)  // [B,S,H*80] f16
{
  __shared__ _Float16 Pls[4 * 16 * 32];       // per-wave P staging tile

  const int lane = threadIdx.x & 31;
  const int wave = threadIdx.x >> 5;
  const int b = blockIdx.z, h = blockIdx.y;
  const int q0 = blockIdx.x * 64 + wave * 16;
  const int hi = lane >> 4, nn = lane & 15;

  const _Float16* Qb = Qp + (size_t)(b * Hc + h) * Sc * HDP;
  const _Float16* Kb = Kp + (size_t)(b * Hc + h) * Sc * HDP;
  const _Float16* Vb = Vt + (size_t)(b * Hc + h) * HDc * Sc;
  _Float16* Pw = Pls + wave * 16 * 32;

  // Q fragments for this wave's 16 rows, kept in registers across the loop.
  v16h qa[3];
#pragma unroll
  for (int j = 0; j < 3; ++j)
    qa[j] = load_a_frag(Qb + (size_t)q0 * HDP + 32 * j, HDP, lane);

  v8f po[5] = {};                  // O accumulator, 16 x 80 per wave
  float mrow[8], lrow[8];          // row stats; row m = r + 8*hi matches C layout
#pragma unroll
  for (int r = 0; r < 8; ++r) { mrow[r] = -1e30f; lrow[r] = 0.f; }

  for (int kc = 0; kc < Sc; kc += 32) {
    // prefetch next chunk of K and V into cache (lowers to global_prefetch_b8)
    if (kc + 32 < Sc) {
      __builtin_prefetch(Kb + (size_t)(kc + 32 + lane) * HDP, 0, 3);
      __builtin_prefetch(Vb + (size_t)(lane * 2) * Sc + kc + 32, 0, 3);
    }

    // ---- scores S = Q K^T over HDP=96 (3 WMMA k-steps), 32 keys (2 n-tiles)
    v8f s0 = {}, s1 = {};
#pragma unroll
    for (int j = 0; j < 3; ++j) {
      v16h b0 = load_b_frag(Kb + (size_t)(kc + 0)  * HDP + 32 * j, HDP, lane);
      v16h b1 = load_b_frag(Kb + (size_t)(kc + 16) * HDP + 32 * j, HDP, lane);
      s0 = wmma_f16(qa[j], b0, s0);
      s1 = wmma_f16(qa[j], b1, s1);
    }

    // ---- online softmax per row
#pragma unroll
    for (int r = 0; r < 8; ++r) {
      float v0 = s0[r] * SCALE_C, v1 = s1[r] * SCALE_C;
      float mx = fmaxf(v0, v1);
#pragma unroll
      for (int off = 1; off < 16; off <<= 1)
        mx = fmaxf(mx, __shfl_xor(mx, off, 32));
      float mnew  = fmaxf(mrow[r], mx);
      float alpha = __expf(mrow[r] - mnew);
      float p0 = __expf(v0 - mnew), p1 = __expf(v1 - mnew);
      float rs = p0 + p1;
#pragma unroll
      for (int off = 1; off < 16; off <<= 1)
        rs += __shfl_xor(rs, off, 32);
      lrow[r] = lrow[r] * alpha + rs;
      mrow[r] = mnew;
#pragma unroll
      for (int t = 0; t < 5; ++t) po[t][r] *= alpha;
      int m = r + 8 * hi;
      Pw[m * 32 + nn]      = (_Float16)p0;
      Pw[m * 32 + 16 + nn] = (_Float16)p1;
    }
    __syncthreads();   // make P tile visible / ordered for the A-layout reads

    // ---- O += P V : A = P (16x32), B = Vt rows (d, s-contiguous)
    v16h pa = load_a_frag(Pw, 32, lane);
#pragma unroll
    for (int t = 0; t < 5; ++t) {
      v16h vb = load_b_frag(Vb + (size_t)(t * 16) * Sc + kc, Sc, lane);
      po[t] = wmma_f16(pa, vb, po[t]);
    }
    __syncthreads();   // protect Pw before next iteration overwrites it
  }

  // ---- finalize: divide by row sums, write O as f16 [B,S,H*80]
#pragma unroll
  for (int r = 0; r < 8; ++r) {
    float inv = 1.0f / lrow[r];
    int s = q0 + r + 8 * hi;
#pragma unroll
    for (int t = 0; t < 5; ++t)
      Oh[((size_t)(b * Sc + s) * Hc + h) * HDc + t * 16 + nn] =
          (_Float16)(po[t][r] * inv);
  }
}

// ---------------------------------------------------------------------------

extern "C" void kernel_launch(void* const* d_in, const int* in_sizes, int n_in,
                              void* d_out, int out_size, void* d_ws, size_t ws_size,
                              hipStream_t stream) {
  (void)in_sizes; (void)n_in; (void)out_size; (void)ws_size;
  const float* x        = (const float*)d_in[0];  // [B,S,DIM]
  const float* rope_cos = (const float*)d_in[1];  // [S,80]
  const float* rope_sin = (const float*)d_in[2];  // [S,80]
  const float* Wqkv     = (const float*)d_in[3];  // [DIM, 3*DIM]
  const float* bqkv     = (const float*)d_in[4];  // [3*DIM]
  const float* Wproj    = (const float*)d_in[5];  // [DIM, DIM]
  const float* bproj    = (const float*)d_in[6];  // [DIM]
  float* out = (float*)d_out;                     // [B,S,DIM] f32

  const int M = Bc * Sc;           // 4096
  const int K = DIMc;              // 1280
  const int N1 = 3 * DIMc;         // 3840
  const int N2 = DIMc;             // 1280

  // workspace carve-up (256B aligned slices)
  char* base = (char*)d_ws;
  size_t off = 0;
  auto carve = [&](size_t bytes) {
    char* p = base + off;
    off += (bytes + 255) & ~(size_t)255;
    return p;
  };
  _Float16* xh     = (_Float16*)carve((size_t)M * K * 2);        // x in f16
  _Float16* WqkvT  = (_Float16*)carve((size_t)N1 * K * 2);       // Wqkv^T f16
  _Float16* WprojT = (_Float16*)carve((size_t)N2 * K * 2);       // Wproj^T f16
  float*    qkvws  = (float*)   carve((size_t)M * N1 * 4);       // qkv f32
  _Float16* Qp     = (_Float16*)carve((size_t)Bc * Hc * Sc * HDP * 2);
  _Float16* Kp     = (_Float16*)carve((size_t)Bc * Hc * Sc * HDP * 2);
  _Float16* Vt     = (_Float16*)carve((size_t)Bc * Hc * HDc * Sc * 2);
  _Float16* Oh     = (_Float16*)carve((size_t)M * K * 2);        // attn out f16

  // 1) convert x to f16
  {
    size_t n = (size_t)M * K;
    cvt_f32_to_f16_kernel<<<dim3((n + 255) / 256), dim3(256), 0, stream>>>(x, xh, n);
  }
  // 2) transpose+convert weights
  {
    size_t n = (size_t)K * N1;
    transpose_cvt_kernel<<<dim3((n + 255) / 256), dim3(256), 0, stream>>>(Wqkv, WqkvT, K, N1);
  }
  {
    size_t n = (size_t)K * N2;
    transpose_cvt_kernel<<<dim3((n + 255) / 256), dim3(256), 0, stream>>>(Wproj, WprojT, K, N2);
  }
  // 3) qkv = x @ Wqkv + bqkv
  gemm_f16_bias_kernel<<<dim3(M / 128, N1 / 128), dim3(256), 0, stream>>>(
      xh, WqkvT, bqkv, qkvws, M, N1, K);
  // 4) RoPE + repack into WMMA-friendly layouts
  {
    size_t n = (size_t)Bc * Hc * Sc * HDP;
    rope_pack_kernel<<<dim3((n + 255) / 256), dim3(256), 0, stream>>>(
        qkvws, rope_cos, rope_sin, Qp, Kp, Vt);
  }
  // 5) flash attention
  attn_kernel<<<dim3(Sc / 64, Hc, Bc), dim3(128), 0, stream>>>(Qp, Kp, Vt, Oh);
  // 6) out = O @ Wproj + bproj
  gemm_f16_bias_kernel<<<dim3(M / 128, N2 / 128), dim3(256), 0, stream>>>(
      Oh, WprojT, bproj, out, M, N2, K);
}
